// MultiAssetSeq2SeqLSTM_65042984731062
// MI455X (gfx1250) — matrix operand order repaired and checked by reading, not compiled, not run on hardware
//
#include <hip/hip_runtime.h>

typedef __bf16 bf16;
typedef __attribute__((ext_vector_type(16))) __bf16 v16bf;
typedef __attribute__((ext_vector_type(8)))  float  v8f;

#define HDIM      512
#define FOURH     2048
#define BATCH     1024
#define SEQ       256
#define INDIM     64
#define NTICK     8
#define HORIZON_N 24

__device__ __forceinline__ float fast_sig(float x) {
    return __builtin_amdgcn_rcpf(1.0f + __expf(-x));
}
__device__ __forceinline__ float fast_tanh(float x) {
    return 1.0f - 2.0f * __builtin_amdgcn_rcpf(1.0f + __expf(2.0f * x));
}

// ---------------- utility kernels ----------------
__global__ void k_f32_to_bf16(const float* __restrict__ src, bf16* __restrict__ dst, int n) {
    int i = blockIdx.x * blockDim.x + threadIdx.x;
    int stride = gridDim.x * blockDim.x;
    for (; i < n; i += stride) dst[i] = (bf16)src[i];
}

__global__ void k_zero_u32(unsigned int* __restrict__ p, int n) {
    int i = blockIdx.x * blockDim.x + threadIdx.x;
    int stride = gridDim.x * blockDim.x;
    for (; i < n; i += stride) p[i] = 0u;
}

// ---------------- fused LSTM cell step ----------------
// gates[b, 0:4H] = A1[b,:] @ W1^T  (+ A2[b,:] @ W2^T)  (+ dot8(dotIn[b,:], dotW))  + bias
// then  c = sig(f)*c + sig(i)*tanh(g);  h_out = sig(o)*tanh(c)   (h_out bf16)
//
// Block tile: 64 batch rows x 64 hidden cols -> 64 x 256 gate columns. 256 threads = 8 waves.
// A tile (8x cross-wave reuse): LDS, double-buffered, pipelined via two scalar uint4
// registers (no arrays/lambda-captured state -> guaranteed SROA, no alloca demotion).
// B (weight) fragments: loaded DIRECTLY from global/L2 -- weights are L2-resident and
// have zero intra-block reuse, so LDS staging would be pure overhead.
template<bool HAS_A2, bool HAS_DOT8>
__global__ __launch_bounds__(256, 1)
void k_lstm_cell(const bf16* __restrict__ A1, int lda1,
                 const bf16* __restrict__ W1, int K1,
                 const bf16* __restrict__ A2, const bf16* __restrict__ W2,   // lda2 = K2 = HDIM
                 const float* __restrict__ dotIn,                             // (B, 8)
                 const float* __restrict__ dotW,                              // (4H, 8) f32
                 const float* __restrict__ bias,                              // (4H)
                 float* __restrict__ cst,                                     // (B, H) f32, in/out
                 bf16* __restrict__ h_out)                                    // (B, H) bf16
{
    // LDS: 2 x A(64x72) bf16 = 18432 B, aliased with 64x260 f32 gate buffer (66560 B)
    __shared__ __align__(16) unsigned char smem[66560];
    bf16 (*As)[64][72] = (bf16(*)[64][72])smem;      // [buf][row][col]
    float (*G)[260]    = (float(*)[260])smem;        // gate staging (aliased)

    const int tid  = threadIdx.x;
    const int wave = tid >> 5;
    const int lane = tid & 31;
    const int m0   = blockIdx.x * 64;   // batch row base
    const int hc0  = blockIdx.y * 64;   // hidden col base

    v8f acc[4][2];
    #pragma unroll
    for (int mi = 0; mi < 4; ++mi)
        #pragma unroll
        for (int ni = 0; ni < 2; ++ni)
            acc[mi][ni] = (v8f)0.0f;

    const int n1  = K1 / 64;
    const int nt_ = n1 + (HAS_A2 ? HDIM / 64 : 0);

    // B-fragment lane mapping: lane = N (0..15), K-half by lane>=16, K packed in regs.
    const int bn    = lane & 15;
    const int khalf = (lane >> 4) * 8;
    const int col0  = wave * 32 + bn;        // local gate column for ni=0
    const int col1  = col0 + 16;             // ni=1
    const int grow0 = (col0 >> 6) * HDIM + hc0 + (col0 & 63);   // W row (i/f/g/o chunked)
    const int grow1 = (col1 >> 6) * HDIM + hc0 + (col1 & 63);

    // A-staging: this thread's slot (4 threads/row, 16 elems each)
    const int sr = tid >> 2, scs = (tid & 3) * 16;

    // source address for this thread's slice of K-tile i (64-deep)
    auto tile_src = [&](int i) -> const uint4* {
        return (i < n1)
            ? (const uint4*)(A1 + (size_t)(m0 + sr) * lda1 + i * 64 + scs)
            : (const uint4*)(A2 + (size_t)(m0 + sr) * HDIM + (i - n1) * 64 + scs);
    };

    uint4 a0, a1;                       // plain scalars: stay in VGPRs
    { const uint4* pa = tile_src(0); a0 = pa[0]; a1 = pa[1]; }
    *(uint4*)&As[0][sr][scs]     = a0;
    *(uint4*)&As[0][sr][scs + 8] = a1;
    __syncthreads();

    for (int kt = 0; kt < nt_; ++kt) {
        const int p = kt & 1;
        const bool more = (kt + 1 < nt_);

        // resolve weight source for this K-tile
        const bf16* W; int K, k0;
        if (kt < n1) { W = W1; K = K1; k0 = kt * 64; }
        else         { W = W2; K = HDIM; k0 = (kt - n1) * 64; }
        const bf16* wr0 = W + (size_t)grow0 * K + k0 + khalf;
        const bf16* wr1 = W + (size_t)grow1 * K + k0 + khalf;

        if (more) { const uint4* pa = tile_src(kt + 1); a0 = pa[0]; a1 = pa[1]; }

        #pragma unroll
        for (int ks = 0; ks < 64; ks += 32) {
            // ---- issue ALL fragment loads first, then the WMMA chain ----
            // B frags straight from global (L2-resident weights)
            v16bf bf0, bf1;
            {
                uint4* q = (uint4*)&bf0;
                q[0] = *(const uint4*)(wr0 + ks);
                q[1] = *(const uint4*)(wr0 + ks + 16);
                uint4* r = (uint4*)&bf1;
                r[0] = *(const uint4*)(wr1 + ks);
                r[1] = *(const uint4*)(wr1 + ks + 16);
            }
            // A frags: lane<16 -> K ks+0..7 / ks+16..23 ; lane>=16 -> ks+8..15 / ks+24..31
            const int arow = lane & 15, ak = ks + (lane >> 4) * 8;
            v16bf af0, af1, af2, af3;
            {
                uint4* q;
                q = (uint4*)&af0;
                q[0] = *(const uint4*)&As[p][arow][ak];
                q[1] = *(const uint4*)&As[p][arow][ak + 16];
                q = (uint4*)&af1;
                q[0] = *(const uint4*)&As[p][16 + arow][ak];
                q[1] = *(const uint4*)&As[p][16 + arow][ak + 16];
                q = (uint4*)&af2;
                q[0] = *(const uint4*)&As[p][32 + arow][ak];
                q[1] = *(const uint4*)&As[p][32 + arow][ak + 16];
                q = (uint4*)&af3;
                q[0] = *(const uint4*)&As[p][48 + arow][ak];
                q[1] = *(const uint4*)&As[p][48 + arow][ak + 16];
            }
            acc[0][0] = __builtin_amdgcn_wmma_f32_16x16x32_bf16(false, af0, false, bf0, (short)0, acc[0][0], false, false);
            acc[0][1] = __builtin_amdgcn_wmma_f32_16x16x32_bf16(false, af0, false, bf1, (short)0, acc[0][1], false, false);
            acc[1][0] = __builtin_amdgcn_wmma_f32_16x16x32_bf16(false, af1, false, bf0, (short)0, acc[1][0], false, false);
            acc[1][1] = __builtin_amdgcn_wmma_f32_16x16x32_bf16(false, af1, false, bf1, (short)0, acc[1][1], false, false);
            acc[2][0] = __builtin_amdgcn_wmma_f32_16x16x32_bf16(false, af2, false, bf0, (short)0, acc[2][0], false, false);
            acc[2][1] = __builtin_amdgcn_wmma_f32_16x16x32_bf16(false, af2, false, bf1, (short)0, acc[2][1], false, false);
            acc[3][0] = __builtin_amdgcn_wmma_f32_16x16x32_bf16(false, af3, false, bf0, (short)0, acc[3][0], false, false);
            acc[3][1] = __builtin_amdgcn_wmma_f32_16x16x32_bf16(false, af3, false, bf1, (short)0, acc[3][1], false, false);
        }

        if (more) {
            *(uint4*)&As[p ^ 1][sr][scs]     = a0;
            *(uint4*)&As[p ^ 1][sr][scs + 8] = a1;
        }
        __syncthreads();
    }

    // gates -> LDS (aliases A-tile buffers; barrier above protects)
    {   // C/D layout: lane<16 -> rows m..m+7 (vgpr e), lane>=16 -> rows m+8..m+15
        int cn = lane & 15;
        int rb = (lane >> 4) * 8;
        #pragma unroll
        for (int mi = 0; mi < 4; ++mi)
            #pragma unroll
            for (int ni = 0; ni < 2; ++ni)
                #pragma unroll
                for (int e = 0; e < 8; ++e)
                    G[mi * 16 + rb + e][wave * 32 + ni * 16 + cn] = acc[mi][ni][e];
    }
    __syncthreads();

    // pointwise LSTM update: 4096 elems / 256 threads
    #pragma unroll 4
    for (int it = 0; it < 16; ++it) {
        int idx = it * 256 + tid;
        int r  = idx >> 6;
        int hc = idx & 63;
        int b    = m0 + r;
        int gcol = hc0 + hc;

        float gi = G[r][hc]        + bias[gcol];
        float gf = G[r][64 + hc]   + bias[HDIM + gcol];
        float gg = G[r][128 + hc]  + bias[2 * HDIM + gcol];
        float go = G[r][192 + hc]  + bias[3 * HDIM + gcol];

        if (HAS_DOT8) {
            float d[NTICK];
            #pragma unroll
            for (int k = 0; k < NTICK; ++k) d[k] = dotIn[(size_t)b * NTICK + k];
            const float* wi = dotW + (size_t)gcol * NTICK;
            const float* wf = dotW + (size_t)(HDIM + gcol) * NTICK;
            const float* wg = dotW + (size_t)(2 * HDIM + gcol) * NTICK;
            const float* wo = dotW + (size_t)(3 * HDIM + gcol) * NTICK;
            #pragma unroll
            for (int k = 0; k < NTICK; ++k) {
                gi = fmaf(d[k], wi[k], gi);
                gf = fmaf(d[k], wf[k], gf);
                gg = fmaf(d[k], wg[k], gg);
                go = fmaf(d[k], wo[k], go);
            }
        }

        size_t ci = (size_t)b * HDIM + gcol;
        float c_old = cst[ci];
        float c_new = fast_sig(gf) * c_old + fast_sig(gi) * fast_tanh(gg);
        cst[ci]   = c_new;
        h_out[ci] = (bf16)(fast_sig(go) * fast_tanh(c_new));
    }
}

// ---------------- output projection (NT=8) + decoder feedback ----------------
__global__ __launch_bounds__(256)
void k_fc_out(const bf16* __restrict__ h1, const float* __restrict__ Wfc,
              const float* __restrict__ bfc, float* __restrict__ out,
              float* __restrict__ di, int t)
{
    int idx = blockIdx.x * blockDim.x + threadIdx.x;   // BATCH*NTICK = 8192
    if (idx >= BATCH * NTICK) return;
    int b = idx >> 3, j = idx & 7;
    const bf16*  hr = h1  + (size_t)b * HDIM;
    const float* wr = Wfc + (size_t)j * HDIM;
    float accv = bfc[j];
    #pragma unroll 8
    for (int k = 0; k < HDIM; ++k) accv = fmaf((float)hr[k], wr[k], accv);
    out[(size_t)b * (HORIZON_N * NTICK) + t * NTICK + j] = accv;
    di[idx] = accv;
}

// ---------------- host orchestration ----------------
extern "C" void kernel_launch(void* const* d_in, const int* in_sizes, int n_in,
                              void* d_out, int out_size, void* d_ws, size_t ws_size,
                              hipStream_t stream) {
    const float* x     = (const float*)d_in[0];
    const float* Wih0e = (const float*)d_in[1];
    const float* Whh0e = (const float*)d_in[2];
    const float* b0e   = (const float*)d_in[3];
    const float* Wih1e = (const float*)d_in[4];
    const float* Whh1e = (const float*)d_in[5];
    const float* b1e   = (const float*)d_in[6];
    const float* Wih0d = (const float*)d_in[7];   // stays f32 (K=8 scalar path)
    const float* Whh0d = (const float*)d_in[8];
    const float* b0d   = (const float*)d_in[9];
    const float* Wih1d = (const float*)d_in[10];
    const float* Whh1d = (const float*)d_in[11];
    const float* b1d   = (const float*)d_in[12];
    const float* Wfc   = (const float*)d_in[13];
    const float* bfc   = (const float*)d_in[14];
    float* outp = (float*)d_out;

    char* ws = (char*)d_ws;
    size_t off = 0;
    auto take = [&](size_t bytes) -> char* {
        char* p = ws + off;
        off = (off + bytes + 255) & ~(size_t)255;
        return p;
    };
    bf16* x_bf     = (bf16*)take((size_t)BATCH * SEQ * INDIM * 2);
    bf16* Wih0e_bf = (bf16*)take((size_t)FOURH * INDIM * 2);
    bf16* Whh0e_bf = (bf16*)take((size_t)FOURH * HDIM * 2);
    bf16* Wih1e_bf = (bf16*)take((size_t)FOURH * HDIM * 2);
    bf16* Whh1e_bf = (bf16*)take((size_t)FOURH * HDIM * 2);
    bf16* Whh0d_bf = (bf16*)take((size_t)FOURH * HDIM * 2);
    bf16* Wih1d_bf = (bf16*)take((size_t)FOURH * HDIM * 2);
    bf16* Whh1d_bf = (bf16*)take((size_t)FOURH * HDIM * 2);
    bf16* h0buf[2] = { (bf16*)take((size_t)BATCH * HDIM * 2), (bf16*)take((size_t)BATCH * HDIM * 2) };
    bf16* h1buf[2] = { (bf16*)take((size_t)BATCH * HDIM * 2), (bf16*)take((size_t)BATCH * HDIM * 2) };
    float* c0 = (float*)take((size_t)BATCH * HDIM * 4);
    float* c1 = (float*)take((size_t)BATCH * HDIM * 4);
    float* di = (float*)take((size_t)BATCH * NTICK * 4);

    // --- precision conversion (once per call; trivial vs 1.9 TFLOP of GEMM) ---
    k_f32_to_bf16<<<2048, 256, 0, stream>>>(x,     x_bf,     BATCH * SEQ * INDIM);
    k_f32_to_bf16<<<512,  256, 0, stream>>>(Wih0e, Wih0e_bf, FOURH * INDIM);
    k_f32_to_bf16<<<2048, 256, 0, stream>>>(Whh0e, Whh0e_bf, FOURH * HDIM);
    k_f32_to_bf16<<<2048, 256, 0, stream>>>(Wih1e, Wih1e_bf, FOURH * HDIM);
    k_f32_to_bf16<<<2048, 256, 0, stream>>>(Whh1e, Whh1e_bf, FOURH * HDIM);
    k_f32_to_bf16<<<2048, 256, 0, stream>>>(Whh0d, Whh0d_bf, FOURH * HDIM);
    k_f32_to_bf16<<<2048, 256, 0, stream>>>(Wih1d, Wih1d_bf, FOURH * HDIM);
    k_f32_to_bf16<<<2048, 256, 0, stream>>>(Whh1d, Whh1d_bf, FOURH * HDIM);

    // --- state init (deterministic every call) ---
    k_zero_u32<<<512, 256, 0, stream>>>((unsigned int*)c0,       BATCH * HDIM);
    k_zero_u32<<<512, 256, 0, stream>>>((unsigned int*)c1,       BATCH * HDIM);
    k_zero_u32<<<512, 256, 0, stream>>>((unsigned int*)h0buf[0], BATCH * HDIM / 2);
    k_zero_u32<<<512, 256, 0, stream>>>((unsigned int*)h1buf[0], BATCH * HDIM / 2);
    k_zero_u32<<<32,  256, 0, stream>>>((unsigned int*)di,       BATCH * NTICK);

    dim3 gridC(BATCH / 64, HDIM / 64);   // 16 x 8 blocks
    int p0 = 0, p1 = 0;

    // --- encoder: 256 steps x 2 layers ---
    for (int t = 0; t < SEQ; ++t) {
        const bf16* xt = x_bf + (size_t)t * INDIM;                 // row stride SEQ*INDIM
        const bf16* h0r = h0buf[p0];  bf16* h0w = h0buf[p0 ^ 1];
        k_lstm_cell<true, false><<<gridC, 256, 0, stream>>>(
            xt, SEQ * INDIM, Wih0e_bf, INDIM, h0r, Whh0e_bf,
            nullptr, nullptr, b0e, c0, h0w);
        const bf16* h1r = h1buf[p1];  bf16* h1w = h1buf[p1 ^ 1];
        k_lstm_cell<true, false><<<gridC, 256, 0, stream>>>(
            h0w, HDIM, Wih1e_bf, HDIM, h1r, Whh1e_bf,
            nullptr, nullptr, b1e, c1, h1w);
        p0 ^= 1; p1 ^= 1;
    }

    // --- decoder: 24 steps x 2 layers + projection feedback ---
    for (int t = 0; t < HORIZON_N; ++t) {
        const bf16* h0r = h0buf[p0];  bf16* h0w = h0buf[p0 ^ 1];
        k_lstm_cell<false, true><<<gridC, 256, 0, stream>>>(
            h0r, HDIM, Whh0d_bf, HDIM, nullptr, nullptr,
            di, Wih0d, b0d, c0, h0w);
        const bf16* h1r = h1buf[p1];  bf16* h1w = h1buf[p1 ^ 1];
        k_lstm_cell<true, false><<<gridC, 256, 0, stream>>>(
            h0w, HDIM, Wih1d_bf, HDIM, h1r, Whh1d_bf,
            nullptr, nullptr, b1d, c1, h1w);
        k_fc_out<<<(BATCH * NTICK + 255) / 256, 256, 0, stream>>>(
            h1w, Wfc, bfc, outp, di, t);
        p0 ^= 1; p1 ^= 1;
    }
}
